// PVMLayer_34643206210208
// MI455X (gfx1250) — matrix operand order repaired and checked by reading, not compiled, not run on hardware
//
#include <hip/hip_runtime.h>
#include <hip/hip_bf16.h>
#include <math.h>

typedef __attribute__((ext_vector_type(16))) _Float16 v16h;
typedef __attribute__((ext_vector_type(8)))  _Float16 v8h;
typedef __attribute__((ext_vector_type(8)))  float    v8f;
typedef __attribute__((ext_vector_type(4))) unsigned int u32x4;
typedef __attribute__((ext_vector_type(8))) int          i32x8;
typedef __attribute__((ext_vector_type(4))) int          i32x4;

#define BATCH  8
#define CCH    256
#define LSEQ   4096
#define DM     64
#define DI     128
#define DSTATE 16
#define NSEQ   32                 // NUM_CHUNKS * BATCH
#define TOK    (NSEQ * LSEQ)      // 131072 mamba tokens
#define TOKB   (BATCH * LSEQ)     // 32768 outer tokens
#define ODIM   256
#define KMAX   256

enum { GEMM_F32 = 0, GEMM_F16 = 1, GEMM_OUT = 2 };

#if defined(__has_builtin)
#if __has_builtin(__builtin_amdgcn_tensor_load_to_lds) && \
    __has_builtin(__builtin_amdgcn_s_wait_tensorcnt)
#define HAVE_TDM 1
#endif
#endif

// ---------------------------------------------------------------------------
// Prep: f32->f16 weight conversion; fold dt_proj into x_proj (combined 160x128
// weight: rows 0..127 = dt_proj @ x_proj[0:4], rows 128..159 = x_proj[4:36]);
// A = -exp(A_log).
// ---------------------------------------------------------------------------
__global__ __launch_bounds__(256) void prep_kernel(
    const float* __restrict__ inproj, const float* __restrict__ xproj,
    const float* __restrict__ dtproj, const float* __restrict__ outproj,
    const float* __restrict__ proj,   const float* __restrict__ Alog,
    _Float16* __restrict__ inproj_h, _Float16* __restrict__ comb_h,
    _Float16* __restrict__ outproj_h, _Float16* __restrict__ proj_h,
    float* __restrict__ Aneg)
{
    int tid = blockIdx.x * 256 + threadIdx.x;
    if (tid < 256 * 64)  inproj_h[tid]  = (_Float16)inproj[tid];
    if (tid < 64 * 128)  outproj_h[tid] = (_Float16)outproj[tid];
    if (tid < 256 * 256) proj_h[tid]    = (_Float16)proj[tid];
    if (tid < 128 * 16)  Aneg[tid]      = -expf(Alog[tid]);
    if (tid < 160 * 128) {
        int r = tid >> 7, k = tid & 127;
        float v;
        if (r < DI) {
            v = 0.f;
            #pragma unroll
            for (int j = 0; j < 4; ++j) v += dtproj[r * 4 + j] * xproj[j * 128 + k];
        } else {
            v = xproj[(4 + (r - DI)) * 128 + k];
        }
        comb_h[tid] = (_Float16)v;
    }
}

// ---------------------------------------------------------------------------
// LayerNorm over C=256 + channel-chunk scatter to (NSEQ, L, 64) f16.
// One wave32 per token; 8 channels per lane; butterfly reduce.
// ---------------------------------------------------------------------------
__global__ __launch_bounds__(256) void ln_chunk_kernel(
    const float* __restrict__ x, const float* __restrict__ g,
    const float* __restrict__ beta, _Float16* __restrict__ xh)
{
    int lane = threadIdx.x & 31;
    int wave = threadIdx.x >> 5;
    int t = blockIdx.x * 8 + wave;
    int b = t >> 12;               // / LSEQ
    int l = t & (LSEQ - 1);
    float v[8], s = 0.f, s2 = 0.f;
    #pragma unroll
    for (int i = 0; i < 8; ++i) {
        int c = lane + 32 * i;
        float val = x[((size_t)b * CCH + c) * LSEQ + l];
        v[i] = val; s += val; s2 += val * val;
    }
    #pragma unroll
    for (int off = 16; off > 0; off >>= 1) {
        s  += __shfl_xor(s,  off, 32);
        s2 += __shfl_xor(s2, off, 32);
    }
    float mu  = s * (1.f / CCH);
    float var = s2 * (1.f / CCH) - mu * mu;
    float r   = rsqrtf(var + 1e-5f);
    #pragma unroll
    for (int i = 0; i < 8; ++i) {
        int c = lane + 32 * i;
        float xn = (v[i] - mu) * r * g[c] + beta[c];
        int chunk = c >> 6, d = c & 63;
        xh[((size_t)(chunk * BATCH + b) * LSEQ + l) * DM + d] = (_Float16)xn;
    }
}

// ---------------------------------------------------------------------------
// WMMA GEMM: out[M,N] = A[M,K] * W[N,K]^T. f16 in, f32 accumulate.
// Block handles one 16-wide N tile shared by 8 waves (one 16-row M tile each).
// The 16xK weight tile (contiguous, W row-major over K) is DMA'd into LDS once
// per block via tensor_load_to_lds (TDM), waited with s_wait_tensorcnt, then
// B fragments are served from LDS (ds_load_b128) while A streams from global.
// mode: 0 = f32 row-major, 1 = f16 row-major, 2 = f32 transposed + bias
//       (final proj epilogue: (B*L,256) -> (B,256,L)).
// ---------------------------------------------------------------------------
__global__ __launch_bounds__(256) void wmma_gemm_kernel(
    const _Float16* __restrict__ A, const _Float16* __restrict__ W,
    void* __restrict__ out, const float* __restrict__ bias,
    int M, int N, int K, int mode)
{
    __shared__ _Float16 wtile[16 * KMAX];

    int lane = threadIdx.x & 31;
    int wave = threadIdx.x >> 5;
    int m0 = (blockIdx.x * 8 + wave) * 16;
    int n0 = blockIdx.y * 16;
    int halflo = (lane < 16);

#ifdef HAVE_TDM
    if (threadIdx.x < 32) {
        // D# group 0: count=1 | lds_addr | global_addr[56:0] | type=2
        unsigned long long gaddr = (unsigned long long)(W + (size_t)n0 * K);
        unsigned lds_off = (unsigned)(size_t)(&wtile[0]);   // low 32 bits = LDS offset
        u32x4 g0;
        g0[0] = 1u;                                          // count=1, user mode
        g0[1] = lds_off;
        g0[2] = (unsigned)(gaddr & 0xFFFFFFFFu);
        g0[3] = (unsigned)((gaddr >> 32) & 0x1FFFFFFu) | 0x80000000u; // type=2
        // D# group 1: 2-D tile, data_size=4B, rows of K/2 dwords, 16 rows.
        unsigned td0 = (unsigned)(K / 2);                    // dwords per row
        i32x8 g1;
        g1[0] = (int)(2u << 16);                             // data_size=2 -> 4 bytes
        g1[1] = (int)((td0 & 0xFFFFu) << 16);                // tensor_dim0[15:0]
        g1[2] = (int)((td0 >> 16) | (16u << 16));            // dim0 hi | tensor_dim1=16
        g1[3] = (int)((td0 & 0xFFFFu) << 16);                // dim1 hi=0 | tile_dim0
        g1[4] = 16;                                          // tile_dim1=16, tile_dim2=0
        g1[5] = (int)td0;                                    // tensor_dim0_stride lo
        g1[6] = 0;                                           // stride hi | dim1_stride lo
        g1[7] = 0;                                           // dim1_stride hi
        i32x4 gz4 = {0, 0, 0, 0};                            // 2-D: groups 2/3 unused
        i32x8 gz8 = {0, 0, 0, 0, 0, 0, 0, 0};
        // 6-arg variant (clang-23 / therock headers): groups 0..3 + extra + cpol
        __builtin_amdgcn_tensor_load_to_lds(g0, g1, gz4, gz4, gz8, 0);
        __builtin_amdgcn_s_wait_tensorcnt(0);
    }
#else
    // Fallback: cooperative global->LDS copy of the 16xK contiguous tile.
    {
        const v8h* src = (const v8h*)(W + (size_t)n0 * K);
        v8h* dst = (v8h*)wtile;
        for (int i = threadIdx.x; i < (16 * K) / 8; i += 256) dst[i] = src[i];
    }
#endif
    __syncthreads();

    // A frag (ISA 7.12.2, 16-bit A 16x32): lane m = lane&15,
    //   VGPR0..3 = K kb..kb+7, VGPR4..7 = K kb+16..kb+23, kb = 8*(lane>=16)
    const _Float16* arow = A + (size_t)(m0 + (lane & 15)) * K + (halflo ? 0 : 8);
    // B frag (K x 16) served from LDS: lane n = lane&15, K kb..kb+7 and
    // kb+16..kb+23, kb = 8*(lane>=16)
    const _Float16* brow = wtile + (lane & 15) * K + (halflo ? 0 : 8);

    v8f acc = {0.f, 0.f, 0.f, 0.f, 0.f, 0.f, 0.f, 0.f};
    for (int k0 = 0; k0 < K; k0 += 32) {
        v8h alo = *(const v8h*)(arow + k0);
        v8h ahi = *(const v8h*)(arow + k0 + 16);
        v8h blo = *(const v8h*)(brow + k0);
        v8h bhi = *(const v8h*)(brow + k0 + 16);
        v16h af = __builtin_shufflevector(alo, ahi, 0,1,2,3,4,5,6,7,8,9,10,11,12,13,14,15);
        v16h bf = __builtin_shufflevector(blo, bhi, 0,1,2,3,4,5,6,7,8,9,10,11,12,13,14,15);
        acc = __builtin_amdgcn_wmma_f32_16x16x32_f16(
            false, af, false, bf, (short)0, acc, false, false);
    }

    // C/D frag: lane col = lane&15, rows = VGPR i + 8*(lane>=16)
    int col   = n0 + (lane & 15);
    int rbase = m0 + (halflo ? 0 : 8);
    if (mode == GEMM_F32) {
        float* o = (float*)out;
        #pragma unroll
        for (int i = 0; i < 8; ++i) o[(size_t)(rbase + i) * N + col] = acc[i];
    } else if (mode == GEMM_F16) {
        _Float16* o = (_Float16*)out;
        #pragma unroll
        for (int i = 0; i < 8; ++i) o[(size_t)(rbase + i) * N + col] = (_Float16)acc[i];
    } else {
        float* o = (float*)out;
        float bv = bias[col];
        #pragma unroll
        for (int i = 0; i < 8; ++i) {
            int row = rbase + i;
            int bb = row >> 12;              // / LSEQ
            int ll = row & (LSEQ - 1);
            o[((size_t)bb * ODIM + col) * LSEQ + ll] = acc[i] + bv;
        }
    }
}

// ---------------------------------------------------------------------------
// Depthwise causal conv (k=4) along L + bias + SiLU; x_in = xz[:, 0:128].
// ---------------------------------------------------------------------------
__global__ __launch_bounds__(256) void conv_silu_kernel(
    const _Float16* __restrict__ xz, const float* __restrict__ cw,
    const float* __restrict__ cb, _Float16* __restrict__ xc)
{
    int tid = blockIdx.x * 256 + threadIdx.x;
    int d  = tid & 127;
    int tl = tid >> 7;                 // token index n*L + l
    int l  = tl & (LSEQ - 1);
    float acc = cb[d];
    #pragma unroll
    for (int i = 0; i < 4; ++i) {
        int ll = l - 3 + i;
        if (ll >= 0)
            acc += cw[d * 4 + i] * (float)xz[(size_t)(tl - l + ll) * 256 + d];
    }
    float sg = 1.f / (1.f + expf(-acc));
    xc[(size_t)tl * DI + d] = (_Float16)(acc * sg);
}

// ---------------------------------------------------------------------------
// Selective scan: one block per sequence, thread = d_inner channel,
// 16-element state in registers; shared B/C staged through LDS per step.
// Fuses +u*D and *silu(z); writes f16 operand for out_proj WMMA GEMM.
// ---------------------------------------------------------------------------
__global__ __launch_bounds__(128) void scan_kernel(
    const float* __restrict__ sdat, const _Float16* __restrict__ xc,
    const _Float16* __restrict__ xz, const float* __restrict__ Aneg,
    const float* __restrict__ dtb, const float* __restrict__ Dp,
    _Float16* __restrict__ ycomb)
{
    __shared__ float sBC[32];
    int d = threadIdx.x;
    int n = blockIdx.x;
    float a[DSTATE], h[DSTATE];
    #pragma unroll
    for (int s = 0; s < DSTATE; ++s) { a[s] = Aneg[d * DSTATE + s]; h[s] = 0.f; }
    float bias = dtb[d], dpar = Dp[d];
    for (int l = 0; l < LSEQ; ++l) {
        size_t t    = (size_t)n * LSEQ + l;
        size_t base = t * 160;
        if (threadIdx.x < 32) sBC[threadIdx.x] = sdat[base + 128 + threadIdx.x];
        __syncthreads();
        float dpre  = sdat[base + d] + bias;
        float delta = (dpre > 20.f) ? dpre : log1pf(expf(dpre));
        float u  = (float)xc[t * DI + d];
        float du = delta * u;
        float y  = 0.f;
        #pragma unroll
        for (int s = 0; s < DSTATE; ++s) {
            float dA = expf(delta * a[s]);
            h[s] = dA * h[s] + du * sBC[s];
            y += h[s] * sBC[DSTATE + s];
        }
        y += u * dpar;
        float z  = (float)xz[t * 256 + DI + d];
        float sg = 1.f / (1.f + expf(-z));
        ycomb[t * DI + d] = (_Float16)(y * z * sg);
        __syncthreads();
    }
}

// ---------------------------------------------------------------------------
// Skip-add + chunk gather back to (B,L,256) + LayerNorm -> f16 for final GEMM.
// ---------------------------------------------------------------------------
__global__ __launch_bounds__(256) void skip_ln_kernel(
    const float* __restrict__ ym, const _Float16* __restrict__ xh,
    const float* __restrict__ g, const float* __restrict__ beta,
    const float* __restrict__ skip, _Float16* __restrict__ xm2)
{
    int lane = threadIdx.x & 31;
    int wave = threadIdx.x >> 5;
    int t = blockIdx.x * 8 + wave;
    int b = t >> 12;
    int l = t & (LSEQ - 1);
    float sk = skip[0];
    float v[8], s = 0.f, s2 = 0.f;
    #pragma unroll
    for (int i = 0; i < 8; ++i) {
        int c = lane + 32 * i;
        int chunk = c >> 6, d = c & 63;
        size_t idx = ((size_t)(chunk * BATCH + b) * LSEQ + l) * DM + d;
        float val = ym[idx] + sk * (float)xh[idx];
        v[i] = val; s += val; s2 += val * val;
    }
    #pragma unroll
    for (int off = 16; off > 0; off >>= 1) {
        s  += __shfl_xor(s,  off, 32);
        s2 += __shfl_xor(s2, off, 32);
    }
    float mu  = s * (1.f / CCH);
    float var = s2 * (1.f / CCH) - mu * mu;
    float r   = rsqrtf(var + 1e-5f);
    #pragma unroll
    for (int i = 0; i < 8; ++i) {
        int c = lane + 32 * i;
        float xn = (v[i] - mu) * r * g[c] + beta[c];
        xm2[((size_t)b * LSEQ + l) * CCH + c] = (_Float16)xn;
    }
}

// ---------------------------------------------------------------------------
extern "C" void kernel_launch(void* const* d_in, const int* in_sizes, int n_in,
                              void* d_out, int out_size, void* d_ws, size_t ws_size,
                              hipStream_t stream)
{
    const float* x        = (const float*)d_in[0];
    const float* ln_g     = (const float*)d_in[1];
    const float* ln_b     = (const float*)d_in[2];
    const float* in_proj  = (const float*)d_in[3];
    const float* conv_w   = (const float*)d_in[4];
    const float* conv_b   = (const float*)d_in[5];
    const float* x_proj   = (const float*)d_in[6];
    const float* dt_proj  = (const float*)d_in[7];
    const float* dt_b     = (const float*)d_in[8];
    const float* A_log    = (const float*)d_in[9];
    const float* D_param  = (const float*)d_in[10];
    const float* out_proj = (const float*)d_in[11];
    const float* proj_w   = (const float*)d_in[12];
    const float* proj_b   = (const float*)d_in[13];
    const float* skip     = (const float*)d_in[14];
    (void)in_sizes; (void)n_in; (void)out_size; (void)ws_size;

    char* wsp = (char*)d_ws;
    size_t off = 0;
    auto alloc = [&](size_t bytes) -> void* {
        off = (off + 255) & ~(size_t)255;
        void* p = wsp + off;
        off += bytes;
        return p;
    };
    _Float16* inproj_h  = (_Float16*)alloc((size_t)256 * 64 * 2);
    _Float16* comb_h    = (_Float16*)alloc((size_t)160 * 128 * 2);
    _Float16* outproj_h = (_Float16*)alloc((size_t)64 * 128 * 2);
    _Float16* proj_h    = (_Float16*)alloc((size_t)256 * 256 * 2);
    float*    Aneg      = (float*)   alloc((size_t)128 * 16 * 4);
    _Float16* xh        = (_Float16*)alloc((size_t)TOK * DM * 2);    // normalized chunks
    _Float16* xz_h      = (_Float16*)alloc((size_t)TOK * 256 * 2);   // in_proj out (x_in|z)
    _Float16* xc_h      = (_Float16*)alloc((size_t)TOK * DI * 2);    // conv+silu out
    float*    sdat      = (float*)   alloc((size_t)TOK * 160 * 4);   // delta_pre|B|C
    _Float16* ycomb     = (_Float16*)alloc((size_t)TOK * DI * 2);    // scan out * silu(z)
    float*    ymb       = (float*)   alloc((size_t)TOK * DM * 4);    // out_proj out
    _Float16* xm2       = (_Float16*)alloc((size_t)TOKB * CCH * 2);  // post-LN2

    // 0) weight prep
    prep_kernel<<<256, 256, 0, stream>>>(in_proj, x_proj, dt_proj, out_proj,
                                         proj_w, A_log,
                                         inproj_h, comb_h, outproj_h, proj_h, Aneg);
    // 1) LN1 + chunk
    ln_chunk_kernel<<<TOKB / 8, 256, 0, stream>>>(x, ln_g, ln_b, xh);
    // 2) in_proj: (TOK,64) x (256,64)^T -> f16 (TOK,256)
    wmma_gemm_kernel<<<dim3(TOK / 128, 256 / 16), 256, 0, stream>>>(
        xh, inproj_h, xz_h, nullptr, TOK, 256, DM, GEMM_F16);
    // 3) depthwise causal conv + SiLU
    conv_silu_kernel<<<(TOK * DI) / 256, 256, 0, stream>>>(xz_h, conv_w, conv_b, xc_h);
    // 4) combined x_proj(+dt_proj): (TOK,128) x (160,128)^T -> f32 (TOK,160)
    wmma_gemm_kernel<<<dim3(TOK / 128, 160 / 16), 256, 0, stream>>>(
        xc_h, comb_h, sdat, nullptr, TOK, 160, DI, GEMM_F32);
    // 5) selective scan + gate
    scan_kernel<<<NSEQ, 128, 0, stream>>>(sdat, xc_h, xz_h, Aneg, dt_b, D_param, ycomb);
    // 6) out_proj: (TOK,128) x (64,128)^T -> f32 (TOK,64)
    wmma_gemm_kernel<<<dim3(TOK / 128, 64 / 16), 256, 0, stream>>>(
        ycomb, outproj_h, ymb, nullptr, TOK, DM, DI, GEMM_F32);
    // 7) skip-add + unchunk + LN2
    skip_ln_kernel<<<TOKB / 8, 256, 0, stream>>>(ymb, xh, ln_g, ln_b, skip, xm2);
    // 8) final proj + bias + transpose to (B,256,L)
    wmma_gemm_kernel<<<dim3(TOKB / 128, ODIM / 16), 256, 0, stream>>>(
        xm2, proj_h, d_out, proj_b, TOKB, ODIM, CCH, GEMM_OUT);
}
